// EAM_17197049053850
// MI455X (gfx1250) — matrix operand order repaired and checked by reading, not compiled, not run on hardware
//
#include <hip/hip_runtime.h>

// ---------------------------------------------------------------------------
// EAM (expectation-attention-like) op for MI455X / gfx1250, wave32 + WMMA bf16
// + Tensor Data Mover staging.
//
// eam[k,i] = sum_p (pem[k,p]/D[p]) * exp(scores[p,i]-m[p]),
// scores[p,q] = sum_c proj[c,p]*pem[c,q],  proj = W@x + bias.
// Softmax stats (m,D) depend only on the contraction index p, so the 512MB
// score tensor is never materialized: stats pass + fused recompute pass.
// ~155 GFLOP vs ~130MB traffic -> matrix-core bound -> bf16 WMMA 16x16x32.
// ---------------------------------------------------------------------------

#define DEVI __device__ __forceinline__

typedef __attribute__((ext_vector_type(16))) __bf16 v16bf;
typedef __attribute__((ext_vector_type(8)))  float  v8f;
typedef __attribute__((ext_vector_type(2)))  __bf16 bf16x2;
typedef __attribute__((ext_vector_type(2)))  float  f32x2;
typedef __attribute__((ext_vector_type(4)))  unsigned int u32x4;
typedef __attribute__((ext_vector_type(8)))  unsigned int u32x8;

constexpr int Bv  = 8;     // batch
constexpr int CIN = 512;   // input channels
constexpr int CI  = 256;   // intermediate channels
constexpr int HW  = 4096;  // spatial positions

union BF16x16 {
  unsigned int u[8];
  v16bf        v;
};

DEVI unsigned short f2bf(float f) {          // native v_cvt_bf16 path
  union { __bf16 h; unsigned short u; } c;
  c.h = (__bf16)f;
  return c.u;
}
DEVI unsigned int pack2(float a, float b) {  // native v_cvt_pk_bf16_f32 path
  f32x2 f; f[0] = a; f[1] = b;
  union { bf16x2 h; unsigned int u; } c;
  c.h = __builtin_convertvector(f, bf16x2);
  return c.u;
}
DEVI void ldg8(BF16x16& f, int h, const unsigned short* p) {  // 8 bf16 = b128
  uint4 t = *(const uint4*)p;
  f.u[h*4+0]=t.x; f.u[h*4+1]=t.y; f.u[h*4+2]=t.z; f.u[h*4+3]=t.w;
}
DEVI v8f wmma_bf16(const BF16x16& A, const BF16x16& B, v8f c) {
  return __builtin_amdgcn_wmma_f32_16x16x32_bf16(
      /*neg_a=*/false, A.v, /*neg_b=*/false, B.v,
      /*c_mod=*/(short)0, c, /*reuse_a=*/false, /*reuse_b=*/false);
}

// A-fragment (16x32 bf16): lane row M = lane&15; K run {16g..16g+15}
// with g = lane>>4 (same k order as B).
DEVI void loadA(BF16x16& A, const unsigned short* row, int k0, int g) {
  ldg8(A, 0, row + k0 + 16*g);
  ldg8(A, 1, row + k0 + 16*g + 8);
}
// B-fragment (32x16 bf16): lane col N = lane&15; K run {16g..16g+15}.
// Source must be laid out [N][K] row-major so the 16 halves are contiguous.
DEVI void loadB(BF16x16& B, const unsigned short* row, int k0, int g) {
  ldg8(B, 0, row + k0 + 16*g);
  ldg8(B, 1, row + k0 + 16*g + 8);
}

// ---------------------------------------------------------------------------
// K1: pem fp32 [B][CI][HW] -> pemT bf16 [B][HW][CI]  (transpose via LDS tile)
// ---------------------------------------------------------------------------
__global__ void pemT_kernel(const float* __restrict__ pem,
                            unsigned short* __restrict__ pemT) {
  __shared__ float tile[32][33];
  int b  = blockIdx.z;
  int c0 = blockIdx.y * 32, q0 = blockIdx.x * 32;
  int tx = threadIdx.x & 31, ty = threadIdx.x >> 5;   // 256 thr: ty = 0..7
  const float* src = pem + ((size_t)b*CI + c0)*HW + q0;
  for (int i = 0; i < 32; i += 8)
    tile[ty+i][tx] = src[(size_t)(ty+i)*HW + tx];
  __syncthreads();
  unsigned short* dst = pemT + ((size_t)b*HW + q0)*CI + c0;
  for (int i = 0; i < 32; i += 8)
    dst[(size_t)(ty+i)*CI + tx] = f2bf(tile[tx][ty+i]);
}

// ---------------------------------------------------------------------------
// K2: proj = W@x + bias, stored transposed: projT bf16 [B][HW(p)][CI(o)].
// Block = 8 waves: one o-tile (16) x eight p-tiles (128). K-loop over CIN.
// x tile (32c x 128p fp32) is DMA'd global->LDS by the Tensor Data Mover with
// pad_enable giving a 129-float row stride (bank spread); fragments are then
// built with strided ds_load_b32 + native bf16 converts.
// ---------------------------------------------------------------------------
__global__ void proj_kernel(const float* __restrict__ x,
                            const float* __restrict__ W,
                            const float* __restrict__ bias,
                            unsigned short* __restrict__ projT) {
  __shared__ float xF[32][129];            // [c][p], TDM-padded rows
  int b = blockIdx.z, o0 = blockIdx.y * 16, pstrip = blockIdx.x * 128;
  int lane = threadIdx.x & 31, wave = threadIdx.x >> 5;
  int g = lane >> 4, n = lane & 15;

  v8f acc = {};
  const float* Wrow = W + (size_t)(o0 + n) * CIN;   // A row for this lane

  for (int c0 = 0; c0 < CIN; c0 += 32) {
    if (wave == 0) {
      // ---- Tensor DMA descriptor: 2D tile, 4B elems, LDS row padding ----
      unsigned long long ga =
          (unsigned long long)(const void*)(x + ((size_t)b*CIN + c0)*HW + pstrip);
      unsigned lds = (unsigned)(size_t)(void*)&xF[0][0];
      u32x4 g0;
      g0[0] = 1u;                                           // count=1 (valid)
      g0[1] = lds;                                          // lds_addr
      g0[2] = (unsigned)ga;                                 // global_addr lo
      g0[3] = ((unsigned)(ga >> 32) & 0x01FFFFFFu)
              | 0x80000000u;                                // ga hi | type=2
      u32x8 g1;
      g1[0] = (2u << 16)        // data_size = 4B
            | (1u << 20)        // pad_enable
            | (6u << 22);       // pad_interval: 128 DWORDs; pad_amount=1 DW
      g1[1] = (unsigned)HW << 16;                           // tensor_dim0 lo16
      g1[2] = ((unsigned)HW >> 16) | ((unsigned)CIN << 16); // dim0 hi | dim1 lo
      g1[3] = ((unsigned)CIN >> 16) | (128u << 16);         // dim1 hi | tile_dim0
      g1[4] = 32u;                                          // tile_dim1 (dim2=0)
      g1[5] = (unsigned)HW;                                 // dim0_stride lo32
      g1[6] = 0u;                                           // stride hi bits
      g1[7] = 0u;
      u32x4 gz = {0u, 0u, 0u, 0u};                          // groups 2/3 unused
      asm volatile("tensor_load_to_lds %0, %1, %2, %3"
                   :: "s"(g0), "s"(g1), "s"(gz), "s"(gz)
                   : "memory");
      __builtin_amdgcn_s_wait_tensorcnt(0);
    }
    __syncthreads();

    // A-frag: W row fp32 -> bf16 (two 8-float runs per lane)
    BF16x16 A;
    {
      const float* w0 = Wrow + c0 + 16*g;
      const float* w1 = Wrow + c0 + 16*g + 8;
      float4 f0 = *(const float4*)w0, f1 = *(const float4*)(w0+4);
      float4 f2 = *(const float4*)w1, f3 = *(const float4*)(w1+4);
      A.u[0]=pack2(f0.x,f0.y); A.u[1]=pack2(f0.z,f0.w);
      A.u[2]=pack2(f1.x,f1.y); A.u[3]=pack2(f1.z,f1.w);
      A.u[4]=pack2(f2.x,f2.y); A.u[5]=pack2(f2.z,f2.w);
      A.u[6]=pack2(f3.x,f3.y); A.u[7]=pack2(f3.z,f3.w);
    }
    // B-frag from LDS tile: VGPR v of lane (g,n) holds K=16g+2v(,+1) @ col n
    BF16x16 Bf;
    {
      int p = wave*16 + n;
      for (int v = 0; v < 8; ++v) {
        int c = 16*g + 2*v;
        Bf.u[v] = pack2(xF[c][p], xF[c+1][p]);
      }
    }
    acc = wmma_bf16(A, Bf, acc);
    __syncthreads();
  }

  // D layout: M = r + 8g (row o), N = n (col p). Fuse bias, store 8 bf16.
  float4 b0 = *(const float4*)(bias + o0 + 8*g);
  float4 b1 = *(const float4*)(bias + o0 + 8*g + 4);
  uint4 st;
  st.x = pack2(acc[0]+b0.x, acc[1]+b0.y);
  st.y = pack2(acc[2]+b0.z, acc[3]+b0.w);
  st.z = pack2(acc[4]+b1.x, acc[5]+b1.y);
  st.w = pack2(acc[6]+b1.z, acc[7]+b1.w);
  int pw = pstrip + wave*16;
  *(uint4*)(projT + ((size_t)b*HW + pw + n)*CI + o0 + 8*g) = st;
}

// ---------------------------------------------------------------------------
// K3: softmax stats. One wave per 16-row p-tile sweeping all 4096 q with WMMA.
// Each lane keeps PRIVATE online (m,d) for its own column subset -> zero
// shuffles in the inner loop; one (m,d)-merge across the 16-lane half at end.
// ---------------------------------------------------------------------------
__global__ void stats_kernel(const unsigned short* __restrict__ projT,
                             const unsigned short* __restrict__ pemT,
                             float* __restrict__ mOut,
                             float* __restrict__ dOut) {
  int b = blockIdx.y;
  int lane = threadIdx.x & 31, wave = threadIdx.x >> 5;
  int g = lane >> 4, n = lane & 15;
  int p0 = (blockIdx.x * 8 + wave) * 16;

  // Preload all 8 A-fragments for this p-tile (K = 256 channels).
  const unsigned short* prow = projT + ((size_t)b*HW + p0 + n) * CI;
  BF16x16 Afr[8];
  for (int cc = 0; cc < 8; ++cc) loadA(Afr[cc], prow, 32*cc, g);

  float m_l[8], d_l[8];
  for (int r = 0; r < 8; ++r) { m_l[r] = -3.0e38f; d_l[r] = 0.f; }

  for (int q0 = 0; q0 < HW; q0 += 16) {
    const unsigned short* brow = pemT + ((size_t)b*HW + q0 + n) * CI;
    if (q0 + 16 < HW) __builtin_prefetch(brow + 16*CI, 0, 1);
    v8f s = {};
    for (int cc = 0; cc < 8; ++cc) {
      BF16x16 Bf; loadB(Bf, brow, 32*cc, g);
      s = wmma_bf16(Afr[cc], Bf, s);
    }
    // lane-private online softmax update (row M = r+8g, this lane's col n)
    for (int r = 0; r < 8; ++r) {
      float v  = s[r];
      float mn = fmaxf(m_l[r], v);
      d_l[r] = d_l[r] * __expf(m_l[r] - mn) + __expf(v - mn);
      m_l[r] = mn;
    }
  }
  // cross-lane (m,d) merge within each 16-lane half-wave
  for (int r = 0; r < 8; ++r) {
    float m = m_l[r], d = d_l[r];
    for (int off = 1; off < 16; off <<= 1) {
      float mo = __shfl_xor(m, off);
      float dd = __shfl_xor(d, off);
      float mn = fmaxf(m, mo);
      d = d * __expf(m - mn) + dd * __expf(mo - mn);
      m = mn;
    }
    m_l[r] = m; d_l[r] = d;
  }
  if (n == 0) {
    for (int r = 0; r < 8; ++r) {
      mOut[(size_t)b*HW + p0 + 8*g + r] = m_l[r];
      dOut[(size_t)b*HW + p0 + 8*g + r] = d_l[r];
    }
  }
}

// ---------------------------------------------------------------------------
// K4: a[k,p] = pem[k,p] / D[p]  -> bf16 [B][CI][HW]
// ---------------------------------------------------------------------------
__global__ void a_kernel(const float* __restrict__ pem,
                         const float* __restrict__ dD,
                         unsigned short* __restrict__ aOut) {
  size_t i = (size_t)blockIdx.x * 256 + threadIdx.x;
  int    p = (int)(i & (HW - 1));
  size_t b = i / ((size_t)CI * HW);
  aOut[i] = f2bf(pem[i] / dD[b*HW + p]);
}

// ---------------------------------------------------------------------------
// K5: fused recompute+exp+second GEMM. Block = 8 waves, one i-tile (16 cols).
// Per 128-wide p-step: wave w recomputes score tile (p_sub = w), applies
// exp(s-m), packs bf16 into shared E[16 i][128 p] (ds_store_b128), then every
// wave accumulates its 2 k-tiles with 4 K=32 WMMAs from E (ds_load_b128).
// ---------------------------------------------------------------------------
__global__ void eam_kernel(const unsigned short* __restrict__ projT,
                           const unsigned short* __restrict__ pemT,
                           const unsigned short* __restrict__ aB,
                           const float* __restrict__ mIn,
                           float* __restrict__ out) {
  __shared__ unsigned short E[16][136];   // [i][p], 272B rows (16B aligned)
  int b = blockIdx.y, i0 = blockIdx.x * 16;
  int lane = threadIdx.x & 31, wave = threadIdx.x >> 5;
  int g = lane >> 4, n = lane & 15;

  // B-fragments of the score GEMM are fixed for the whole kernel (cols i0+n).
  const unsigned short* brow = pemT + ((size_t)b*HW + i0 + n) * CI;
  BF16x16 Bs[8];
  for (int cc = 0; cc < 8; ++cc) loadB(Bs[cc], brow, 32*cc, g);

  const unsigned short* prow0 = projT + (size_t)b*HW*CI;
  const unsigned short* arow0 = aB + ((size_t)b*CI + wave*32 + n) * HW;
  const unsigned short* arow1 = arow0 + (size_t)16 * HW;
  const float* mrow = mIn + (size_t)b*HW;

  v8f acc0 = {}, acc1 = {};

  for (int p0 = 0; p0 < HW; p0 += 128) {
    int pw = p0 + wave*16;                    // this wave's score rows
    // --- recompute 16x16 score tile ---
    v8f s = {};
    const unsigned short* prw = prow0 + (size_t)(pw + n) * CI;
    for (int cc = 0; cc < 8; ++cc) {
      BF16x16 A; loadA(A, prw, 32*cc, g);
      s = wmma_bf16(A, Bs[cc], s);
    }
    // --- exp(s - m[p]) -> bf16 -> LDS (rows p = pw + 8g + r) ---
    float4 m0 = *(const float4*)(mrow + pw + 8*g);
    float4 m1 = *(const float4*)(mrow + pw + 8*g + 4);
    uint4 pk;
    pk.x = pack2(__expf(s[0]-m0.x), __expf(s[1]-m0.y));
    pk.y = pack2(__expf(s[2]-m0.z), __expf(s[3]-m0.w));
    pk.z = pack2(__expf(s[4]-m1.x), __expf(s[5]-m1.y));
    pk.w = pack2(__expf(s[6]-m1.z), __expf(s[7]-m1.w));
    *(uint4*)&E[n][wave*16 + 8*g] = pk;
    __syncthreads();
    // --- eam accumulation: 2 k-tiles x 4 K=32 chunks ---
    for (int t = 0; t < 4; ++t) {
      int pl = 32 * t;
      BF16x16 Bf;
      const unsigned short* er = &E[n][pl + 16*g];
      ldg8(Bf, 0, er);
      ldg8(Bf, 1, er + 8);
      BF16x16 A0; loadA(A0, arow0 + p0, pl, g);
      acc0 = wmma_bf16(A0, Bf, acc0);
      BF16x16 A1; loadA(A1, arow1 + p0, pl, g);
      acc1 = wmma_bf16(A1, Bf, acc1);
    }
    __syncthreads();
  }

  // store fp32: k = wave*32 + {0,16} + 8g + r, i = i0 + n
  float* o0 = out + ((size_t)b*CI + wave*32 + 8*g) * HW + i0 + n;
  float* o1 = o0 + (size_t)16 * HW;
  for (int r = 0; r < 8; ++r) {
    o0[(size_t)r * HW] = acc0[r];
    o1[(size_t)r * HW] = acc1[r];
  }
}

// ---------------------------------------------------------------------------
// Workspace layout: projT(16MB) | pemT(16MB) | a(16MB) | m(128KB) | D(128KB)
// ---------------------------------------------------------------------------
extern "C" void kernel_launch(void* const* d_in, const int* in_sizes, int n_in,
                              void* d_out, int out_size, void* d_ws, size_t ws_size,
                              hipStream_t stream) {
  const float* x    = (const float*)d_in[0];
  const float* pem  = (const float*)d_in[1];
  const float* W    = (const float*)d_in[2];
  const float* bias = (const float*)d_in[3];
  float* out = (float*)d_out;

  const size_t SZ = (size_t)Bv * HW * CI * sizeof(unsigned short);  // 16 MiB
  unsigned short* projT = (unsigned short*)d_ws;
  unsigned short* pemT  = (unsigned short*)((char*)d_ws + SZ);
  unsigned short* aB    = (unsigned short*)((char*)d_ws + 2*SZ);
  float* mP = (float*)((char*)d_ws + 3*SZ);
  float* dP = mP + (size_t)Bv * HW;

  {
    dim3 grid(HW/32, CI/32, Bv);
    pemT_kernel<<<grid, 256, 0, stream>>>(pem, pemT);
  }
  {
    dim3 grid(HW/128, CI/16, Bv);
    proj_kernel<<<grid, 256, 0, stream>>>(x, W, bias, projT);
  }
  {
    dim3 grid(HW/128, Bv);
    stats_kernel<<<grid, 256, 0, stream>>>(projT, pemT, mP, dP);
  }
  {
    int total = Bv * CI * HW;
    a_kernel<<<total/256, 256, 0, stream>>>(pem, dP, aB);
  }
  {
    dim3 grid(HW/16, Bv);
    eam_kernel<<<grid, 256, 0, stream>>>(projT, pemT, aB, mP, out);
  }
}
